// ClimatePINN_58093727645965
// MI455X (gfx1250) — matrix operand order repaired and checked by reading, not compiled, not run on hardware
//
#include <hip/hip_runtime.h>
#include <hip/hip_bf16.h>

// ---------------------------------------------------------------------------
// ClimatePINN on gfx1250: WMMA implicit-GEMM convs (NHWC f16, f32 accum) with
// TDM (tensor_load_to_lds) weight staging, coord-MLP via WMMA GEMM, BN via
// LDS reductions, physics via central differences of 7 out_fn evaluations.
// ---------------------------------------------------------------------------

typedef _Float16 half8_t  __attribute__((ext_vector_type(8)));
typedef _Float16 half16_t __attribute__((ext_vector_type(16)));
typedef float    floatx8  __attribute__((ext_vector_type(8)));
typedef int      intx8    __attribute__((ext_vector_type(8)));
typedef unsigned int v4u  __attribute__((ext_vector_type(4)));
typedef int          v8i  __attribute__((ext_vector_type(8)));
typedef int          v4i  __attribute__((ext_vector_type(4)));

#define NPIX 32768   // B*H*W = 16*32*64
#define HH   32
#define WW   64

#if __has_builtin(__builtin_amdgcn_tensor_load_to_lds) && \
    __has_builtin(__builtin_amdgcn_s_wait_tensorcnt)
#define CPINN_HAS_TDM 1
#endif

__device__ inline half16_t h16_zero() {
    intx8 z = {0, 0, 0, 0, 0, 0, 0, 0};
    return __builtin_bit_cast(half16_t, z);
}

// ---------------------------------------------------------------------------
// Weight packer: emit f16 B-fragments in this file's WMMA lane order (contiguous k).
// dst linear index = ((nt*taps + tap)*nck + ckid)*512 + lane*16 + e
//   column n = nt*16 + (lane&15); k = ckid*32 + (lane>>4)*16 + e
// src element (cout=n, cin=k, tap): addr = n*so + k*si + tap*st
// ---------------------------------------------------------------------------
__global__ void cpinn_pack_w_kernel(const float* __restrict__ src,
                                    long so, long si, long st,
                                    int Cin, int Cout, int taps, int ntiles,
                                    _Float16* __restrict__ dst) {
    int nck = Cin >> 5;
    long total = (long)ntiles * taps * nck * 512;
    long idx = (long)blockIdx.x * 256 + threadIdx.x;
    if (idx >= total) return;
    int e    = (int)(idx & 15);
    int lane = (int)((idx >> 4) & 31);
    long rest = idx >> 9;
    int ckid = (int)(rest % nck); rest /= nck;
    int tap  = (int)(rest % taps); rest /= taps;
    int nt   = (int)rest;
    int n = nt * 16 + (lane & 15);
    int k = ckid * 32 + (lane >> 4) * 16 + e;
    float v = (n < Cout) ? src[(long)n * so + (long)k * si + (long)tap * st] : 0.0f;
    dst[idx] = (_Float16)v;
}

// ---------------------------------------------------------------------------
// WMMA implicit-GEMM conv, templated so all index math folds to shifts and
// the tap/K loops fully unroll into straight-line v_wmma chains.
//   - One block = 8 waves, all sharing one N-tile group of NTW tiles.
//   - Block's packed weights (contiguous, nt-major) staged into LDS by the
//     Tensor Data Mover (1-D D#, 8-byte elements), s_wait_tensorcnt + barrier.
//   - Each wave: 16-pixel x (NTW*16)-cout tile; A-fragment shared across NTW.
// in: NHWC f16 stride CIN; out: NHWC f32 stride COUT. K3=1: 3x3 SAME, 0: 1x1.
// ---------------------------------------------------------------------------
template <int CIN, int COUT, int NTW, int K3>
__global__ __launch_bounds__(256)
void cpinn_conv_wmma_t(const _Float16* __restrict__ in,
                       const _Float16* __restrict__ wpk,
                       float* __restrict__ out) {
    constexpr int NCK    = CIN / 32;
    constexpr int TAPS   = K3 ? 9 : 1;
    constexpr int CT     = COUT / 16;
    constexpr int NG     = CT / NTW;                    // n-tile groups
    constexpr int WBYTES = NTW * TAPS * NCK * 1024;     // staged weight bytes

    extern __shared__ _Float16 shw[];

    const int lane = threadIdx.x & 31;
    const int wid  = threadIdx.x >> 5;
    const int ng   = (NG == 1) ? 0 : ((int)blockIdx.x % NG);
    const int mtg  = (NG == 1) ? (int)blockIdx.x : ((int)blockIdx.x / NG);

    // ---- stage this block's weight fragments into LDS (TDM async DMA) ----
    const _Float16* wsrc = wpk + (long)ng * NTW * TAPS * NCK * 512;
#ifdef CPINN_HAS_TDM
    if (threadIdx.x < 32) {   // wave 0 issues one 1-D tensor tile copy
        unsigned long long ga = (unsigned long long)wsrc;
        unsigned int lds = (unsigned int)(unsigned long long)shw;
        unsigned int n8  = WBYTES / 8;              // 8-byte elements
        v4u g0;
        g0.x = 1u;                                  // count=1 (valid D#)
        g0.y = lds;                                 // lds_addr
        g0.z = (unsigned int)ga;                    // global_addr[31:0]
        g0.w = (unsigned int)((ga >> 32) & 0x01ffffffu) | (2u << 30); // type=2
        v8i g1;
        g1[0] = (int)(3u << 16);                    // data_size=8B, mask=0
        g1[1] = (int)((n8 & 0xffffu) << 16);        // tensor_dim0[15:0]
        g1[2] = (int)(((n8 >> 16) & 0xffffu) | (1u << 16)); // dim0 hi | dim1=1
        g1[3] = (int)((n8 & 0xffffu) << 16);        // dim1 hi=0 | tile_dim0=n8
        g1[4] = 1;                                  // tile_dim1=1, tile_dim2=0
        g1[5] = (int)n8;                            // tensor_dim0_stride
        g1[6] = 0;
        g1[7] = 0;
        v4i gz = {0, 0, 0, 0};
#if __clang_major__ >= 23
        v8i z8 = {0, 0, 0, 0, 0, 0, 0, 0};
        __builtin_amdgcn_tensor_load_to_lds(g0, g1, gz, gz, z8, 0);
#else
        __builtin_amdgcn_tensor_load_to_lds(g0, g1, gz, gz, 0);
#endif
        __builtin_amdgcn_s_wait_tensorcnt(0);
    }
    __syncthreads();
#else
    for (int i = threadIdx.x; i < WBYTES / 16; i += 256)
        ((float4*)shw)[i] = ((const float4*)wsrc)[i];
    __syncthreads();
#endif

    const int mt = mtg * 8 + wid;
    const int p0 = mt * 16;
    const int w0 = p0 & 63;
    const int h0 = (p0 >> 6) & 31;
    const int b  = p0 >> 11;
    const int m  = lane & 15;
    const int kh = lane >> 4;

    floatx8 acc[NTW];
#pragma unroll
    for (int j = 0; j < NTW; ++j)
        acc[j] = (floatx8){0.f, 0.f, 0.f, 0.f, 0.f, 0.f, 0.f, 0.f};

    const _Float16* shl = shw + lane * 16;

#pragma unroll
    for (int tap = 0; tap < TAPS; ++tap) {
        const int dy = K3 ? (tap / 3) - 1 : 0;
        const int dx = K3 ? (tap % 3) - 1 : 0;
        const int hs = h0 + dy;
        const int ws = w0 + m + dx;
        const bool ok = (hs >= 0) && (hs < HH) && (ws >= 0) && (ws < WW);
        const _Float16* ap = in + (long)((b * HH + hs) * WW + ws) * CIN + kh * 16;
#pragma unroll
        for (int ckid = 0; ckid < NCK; ++ckid) {
            half16_t a;
            if (ok) {
                half8_t lo = *(const half8_t*)(ap + ckid * 32);      // K 0-7 (+16 hi lanes)
                half8_t hi = *(const half8_t*)(ap + ckid * 32 + 8);  // K 8-15 (+16)
                a = __builtin_shufflevector(lo, hi, 0, 1, 2, 3, 4, 5, 6, 7,
                                                    8, 9, 10, 11, 12, 13, 14, 15);
            } else {
                a = h16_zero();                                      // SAME padding
            }
#pragma unroll
            for (int j = 0; j < NTW; ++j) {
                half16_t bf =
                    *(const half16_t*)(shl + (long)((j * TAPS + tap) * NCK + ckid) * 512);
                acc[j] = __builtin_amdgcn_wmma_f32_16x16x32_f16(
                    false, a, false, bf, (short)0, acc[j], false, false);
            }
        }
    }
    // C/D layout: VGPR r -> M = r + 8*(lane>=16), N = lane&15
    const int n  = lane & 15;
    const int mb = (lane >> 4) * 8;
#pragma unroll
    for (int j = 0; j < NTW; ++j) {
        float* op = out + (long)p0 * COUT + (ng * NTW + j) * 16 + n;
#pragma unroll
        for (int r = 0; r < 8; ++r)
            op[(long)(mb + r) * COUT] = acc[j][r];
    }
}

// ---------------------------------------------------------------------------
// Direct conv for tiny Cin (2/3): input NCHW f32 -> NHWC f32 (C=64).
// ---------------------------------------------------------------------------
__global__ void cpinn_conv_direct_kernel(const float* __restrict__ in,
                                         const float* __restrict__ w,
                                         float* __restrict__ out, int Cin) {
    int idx = blockIdx.x * 256 + threadIdx.x;
    if (idx >= NPIX * 64) return;
    int c = idx & 63, p = idx >> 6;
    int x = p & 63, h = (p >> 6) & 31, b = p >> 11;
    float s = 0.f;
    for (int ci = 0; ci < Cin; ++ci)
        for (int ky = 0; ky < 3; ++ky) {
            int hs = h + ky - 1;
            if (hs < 0 || hs >= HH) continue;
            for (int kx = 0; kx < 3; ++kx) {
                int xs = x + kx - 1;
                if (xs < 0 || xs >= WW) continue;
                s += in[((long)(b * Cin + ci) * HH + hs) * WW + xs] *
                     w[((c * Cin + ci) * 3 + ky) * 3 + kx];
            }
        }
    out[(long)p * 64 + c] = s;
}

// ---------------------------------------------------------------------------
// BatchNorm statistics: one block per channel, LDS tree reduction.
// ---------------------------------------------------------------------------
__global__ __launch_bounds__(256)
void cpinn_bn_stats_kernel(const float* __restrict__ x, int C,
                           float* __restrict__ mean, float* __restrict__ var) {
    __shared__ float ss[256], sq[256];
    int c = blockIdx.x;
    float s = 0.f, q = 0.f;
    for (int p = threadIdx.x; p < NPIX; p += 256) {
        float v = x[(long)p * C + c];
        s += v; q += v * v;
    }
    ss[threadIdx.x] = s; sq[threadIdx.x] = q;
    __syncthreads();
    for (int st = 128; st > 0; st >>= 1) {
        if ((int)threadIdx.x < st) {
            ss[threadIdx.x] += ss[threadIdx.x + st];
            sq[threadIdx.x] += sq[threadIdx.x + st];
        }
        __syncthreads();
    }
    if (threadIdx.x == 0) {
        float m = ss[0] * (1.f / NPIX);
        mean[c] = m;
        var[c]  = sq[0] * (1.f / NPIX) - m * m;
    }
}

// ---------------------------------------------------------------------------
// BN apply + activation. mode: 0=tanh, 1=relu, 2=relu(x+resid), 3=identity.
// Writes f32 (optional) and f16 (into dst tensor with channel offset).
// ---------------------------------------------------------------------------
__global__ void cpinn_bn_apply_kernel(const float* __restrict__ x, int C,
                                      const float* __restrict__ g,
                                      const float* __restrict__ bta,
                                      const float* __restrict__ mean,
                                      const float* __restrict__ var,
                                      float* __restrict__ f32out,
                                      _Float16* __restrict__ f16out,
                                      int dstC, int dstOff, int mode,
                                      const float* __restrict__ resid) {
    long idx = (long)blockIdx.x * 256 + threadIdx.x;
    if (idx >= (long)NPIX * C) return;
    int c = (int)(idx % C);
    long p = idx / C;
    float v = (x[idx] - mean[c]) * rsqrtf(var[c] + 1e-5f) * g[c] + bta[c];
    if (mode == 0)       v = tanhf(v);
    else if (mode == 1)  v = fmaxf(v, 0.f);
    else if (mode == 2)  v = fmaxf(v + resid[idx], 0.f);
    if (f32out) f32out[idx] = v;
    f16out[p * dstC + dstOff + c] = (_Float16)v;
}

// ---------------------------------------------------------------------------
// Coord MLP layer 1: h1 = tanh([x+ex, y+ey, t+et] @ w1 + b1)  (3 -> 64)
// ---------------------------------------------------------------------------
__global__ void cpinn_mlp_h1_kernel(const float* __restrict__ x,
                                    const float* __restrict__ y,
                                    const float* __restrict__ t,
                                    const float* __restrict__ w1,
                                    const float* __restrict__ b1,
                                    float ex, float ey, float et,
                                    _Float16* __restrict__ h1) {
    int idx = blockIdx.x * 256 + threadIdx.x;
    if (idx >= NPIX * 64) return;
    int j = idx & 63, p = idx >> 6;
    int b = p >> 11;
    float xv = x[p] + ex, yv = y[p] + ey, tv = t[b] + et;
    float v = tanhf(xv * w1[j] + yv * w1[64 + j] + tv * w1[128 + j] + b1[j]);
    h1[(long)p * 64 + j] = (_Float16)v;
}

// cf = tanh(h1@w2 + b2), written into channels [128,192) of the concat buffer
__global__ void cpinn_mlp_out_kernel(const float* __restrict__ h2conv,
                                     const float* __restrict__ b2,
                                     _Float16* __restrict__ comb) {
    int idx = blockIdx.x * 256 + threadIdx.x;
    if (idx >= NPIX * 64) return;
    int j = idx & 63, p = idx >> 6;
    comb[(long)p * 192 + 128 + j] = (_Float16)tanhf(h2conv[idx] + b2[j]);
}

// final conv output (NHWC C=16 padded) + bias -> NCHW 3-channel f32
__global__ void cpinn_final_out_kernel(const float* __restrict__ conv,
                                       const float* __restrict__ c2b,
                                       float* __restrict__ out) {
    int idx = blockIdx.x * 256 + threadIdx.x;
    if (idx >= NPIX * 3) return;
    int co = idx % 3;
    int p  = idx / 3;
    int s = p & 2047, b = p >> 11;
    out[(long)b * 6144 + (long)co * 2048 + s] = conv[(long)p * 16 + co] + c2b[co];
}

__global__ void cpinn_zero3_kernel(float* a) {
    if (threadIdx.x < 3) a[threadIdx.x] = 0.f;
}

// ---------------------------------------------------------------------------
// Physics residuals via central differences; LDS reduce + atomicAdd of means.
// ---------------------------------------------------------------------------
__global__ __launch_bounds__(256)
void cpinn_physics_kernel(const float* __restrict__ oc,
                          const float* __restrict__ oxp, const float* __restrict__ oxm,
                          const float* __restrict__ oyp, const float* __restrict__ oym,
                          const float* __restrict__ otp, const float* __restrict__ otm,
                          const float* __restrict__ log_re,
                          float epsS, float epsT, float* __restrict__ accum) {
    int p = blockIdx.x * 256 + threadIdx.x;  // < 32768
    int s = p & 2047, b = p >> 11;
    long i1 = (long)b * 6144 + 2048 + s;   // channel 1 (u)
    long i2 = i1 + 2048;                   // channel 2 (v)
    float u = oc[i1], v = oc[i2];
    float inv2s = 0.5f / epsS, inv2t = 0.5f / epsT, invs2 = 1.f / (epsS * epsS);
    float ux = (oxp[i1] - oxm[i1]) * inv2s, uy = (oyp[i1] - oym[i1]) * inv2s;
    float vx = (oxp[i2] - oxm[i2]) * inv2s, vy = (oyp[i2] - oym[i2]) * inv2s;
    float ut = (otp[i1] - otm[i1]) * inv2t, vt = (otp[i2] - otm[i2]) * inv2t;
    float uxx = (oxp[i1] - 2.f * u + oxm[i1]) * invs2;
    float uyy = (oyp[i1] - 2.f * u + oym[i1]) * invs2;
    float vxx = (oxp[i2] - 2.f * v + oxm[i2]) * invs2;
    float vyy = (oyp[i2] - 2.f * v + oym[i2]) * invs2;
    float lr = fminf(fmaxf(log_re[0], 3.9120230054f), 11.5129254650f);
    float invre = expf(-lr);
    float e1 = ux + uy;
    float e2 = ut + (u * ux + v * uy) - (uxx + uyy) * invre;
    float e3 = vt + (u * vx + v * vy) - (vxx + vyy) * invre;
    __shared__ float sh[3][256];
    sh[0][threadIdx.x] = e1 * e1;
    sh[1][threadIdx.x] = e2 * e2;
    sh[2][threadIdx.x] = e3 * e3;
    __syncthreads();
    for (int st = 128; st > 0; st >>= 1) {
        if ((int)threadIdx.x < st)
            for (int k = 0; k < 3; ++k)
                sh[k][threadIdx.x] += sh[k][threadIdx.x + st];
        __syncthreads();
    }
    if (threadIdx.x < 3)
        atomicAdd(&accum[threadIdx.x], sh[threadIdx.x][0] * (1.f / NPIX));
}

__global__ void cpinn_copy3_kernel(const float* a, float* o) {
    if (threadIdx.x < 3) o[threadIdx.x] = a[threadIdx.x];
}

// ---------------------------------------------------------------------------
// Host orchestration
// ---------------------------------------------------------------------------
extern "C" void kernel_launch(void* const* d_in, const int* in_sizes, int n_in,
                              void* d_out, int out_size, void* d_ws, size_t ws_size,
                              hipStream_t stream) {
    (void)in_sizes; (void)n_in; (void)out_size; (void)ws_size;

    // -- inputs (pytree order: top-level dict order, params dict sorted keys) --
    const float* meteo_in = (const float*)d_in[0];
    const float* masks_in = (const float*)d_in[1];
    const float* x_in     = (const float*)d_in[2];
    const float* y_in     = (const float*)d_in[3];
    const float* t_in     = (const float*)d_in[4];
    const float* cmb_b1  = (const float*)d_in[5];
    const float* cmb_c1w = (const float*)d_in[7];
    const float* cmb_c2b = (const float*)d_in[8];
    const float* cmb_c2w = (const float*)d_in[9];
    const float* cmb_g1  = (const float*)d_in[10];
    const float* cr_b1  = (const float*)d_in[11];
    const float* cr_b2  = (const float*)d_in[12];
    const float* cr_c1w = (const float*)d_in[14];
    const float* cr_c2w = (const float*)d_in[16];
    const float* cr_g1  = (const float*)d_in[17];
    const float* cr_g2  = (const float*)d_in[18];
    const float* co_b1 = (const float*)d_in[19];
    const float* co_b2 = (const float*)d_in[20];
    const float* co_w1 = (const float*)d_in[21];
    const float* co_w2 = (const float*)d_in[22];
    const float* log_re = (const float*)d_in[23];
    struct EncP { const float *b1,*b2,*c1w,*c2w,*g1,*g2,*rb1,*rb2,*rc1w,*rc2w,*rg1,*rg2; };
    auto encAt = [&](int base) {
        EncP e;
        e.b1   = (const float*)d_in[base + 0];
        e.b2   = (const float*)d_in[base + 1];
        e.c1w  = (const float*)d_in[base + 3];
        e.c2w  = (const float*)d_in[base + 5];
        e.g1   = (const float*)d_in[base + 6];
        e.g2   = (const float*)d_in[base + 7];
        e.rb1  = (const float*)d_in[base + 8];
        e.rb2  = (const float*)d_in[base + 9];
        e.rc1w = (const float*)d_in[base + 11];
        e.rc2w = (const float*)d_in[base + 13];
        e.rg1  = (const float*)d_in[base + 14];
        e.rg2  = (const float*)d_in[base + 15];
        return e;
    };
    EncP maskE  = encAt(24);
    EncP meteoE = encAt(40);

    float* out_f = (float*)d_out;

    // -- workspace allocator --
    char* base = (char*)d_ws;
    size_t off = 0;
    auto alloc = [&](size_t bytes) -> void* {
        off = (off + 255) & ~(size_t)255;
        void* p = base + off;
        off += bytes;
        return p;
    };
    const long P = NPIX;
    _Float16* comb_in = (_Float16*)alloc(P * 192 * 2);   // [mf | kf | cf] NHWC f16
    float*    convout = (float*)alloc(P * 128 * 4);      // conv accumulator
    float*    actA32  = (float*)alloc(P * 128 * 4);      // residual-branch save
    _Float16* actA16  = (_Float16*)alloc(P * 128 * 2);
    _Float16* actB16  = (_Float16*)alloc(P * 128 * 2);
    _Float16* actC16  = (_Float16*)alloc(P * 128 * 2);
    _Float16* h1f16   = (_Float16*)alloc(P * 64 * 2);
    float*    meanb   = (float*)alloc(256 * 4);
    float*    varb    = (float*)alloc(256 * 4);
    float*    accum   = (float*)alloc(16 * 4);
    float* oxp = (float*)alloc(P * 3 * 4);
    float* oxm = (float*)alloc(P * 3 * 4);
    float* oyp = (float*)alloc(P * 3 * 4);
    float* oym = (float*)alloc(P * 3 * 4);
    float* otp = (float*)alloc(P * 3 * 4);
    float* otm = (float*)alloc(P * 3 * 4);
    _Float16* pk_me_rc1 = (_Float16*)alloc(4 * 9 * 2 * 512 * 2);
    _Float16* pk_me_rc2 = (_Float16*)alloc(4 * 9 * 2 * 512 * 2);
    _Float16* pk_me_c2  = (_Float16*)alloc(4 * 9 * 2 * 512 * 2);
    _Float16* pk_ma_rc1 = (_Float16*)alloc(4 * 9 * 2 * 512 * 2);
    _Float16* pk_ma_rc2 = (_Float16*)alloc(4 * 9 * 2 * 512 * 2);
    _Float16* pk_ma_c2  = (_Float16*)alloc(4 * 9 * 2 * 512 * 2);
    _Float16* pk_cc1  = (_Float16*)alloc(8 * 9 * 6 * 512 * 2);
    _Float16* pk_crc1 = (_Float16*)alloc(8 * 9 * 4 * 512 * 2);
    _Float16* pk_crc2 = (_Float16*)alloc(8 * 9 * 4 * 512 * 2);
    _Float16* pk_cc2  = (_Float16*)alloc(1 * 9 * 4 * 512 * 2);
    _Float16* pk_w2   = (_Float16*)alloc(4 * 1 * 2 * 512 * 2);

    auto pack = [&](const float* src, long so, long si, long st,
                    int Cin, int Cout, int taps, _Float16* dst) {
        int ntiles = (Cout + 15) / 16;
        long total = (long)ntiles * taps * (Cin >> 5) * 512;
        cpinn_pack_w_kernel<<<(int)((total + 255) / 256), 256, 0, stream>>>(
            src, so, si, st, Cin, Cout, taps, ntiles, dst);
    };
    pack(meteoE.rc1w, 64 * 9, 9, 1, 64, 64, 9, pk_me_rc1);
    pack(meteoE.rc2w, 64 * 9, 9, 1, 64, 64, 9, pk_me_rc2);
    pack(meteoE.c2w,  64 * 9, 9, 1, 64, 64, 9, pk_me_c2);
    pack(maskE.rc1w,  64 * 9, 9, 1, 64, 64, 9, pk_ma_rc1);
    pack(maskE.rc2w,  64 * 9, 9, 1, 64, 64, 9, pk_ma_rc2);
    pack(maskE.c2w,   64 * 9, 9, 1, 64, 64, 9, pk_ma_c2);
    pack(cmb_c1w, 192 * 9, 9, 1, 192, 128, 9, pk_cc1);
    pack(cr_c1w,  128 * 9, 9, 1, 128, 128, 9, pk_crc1);
    pack(cr_c2w,  128 * 9, 9, 1, 128, 128, 9, pk_crc2);
    pack(cmb_c2w, 128 * 9, 9, 1, 128, 3,   9, pk_cc2);
    pack(co_w2, 1, 64, 0, 64, 64, 1, pk_w2);

    // dynamic-LDS sizes = staged weight bytes per block
    constexpr unsigned WB_enc = 2 * 9 * 2 * 1024;   // <64,64,2,1>  36864
    constexpr unsigned WB_mlp = 2 * 1 * 2 * 1024;   // <64,64,2,0>   4096
    constexpr unsigned WB_cc1 = 1 * 9 * 6 * 1024;   // <192,128,1,1> 55296
    constexpr unsigned WB_crc = 1 * 9 * 4 * 1024;   // <128,128,1,1> 36864
    constexpr unsigned WB_fin = 1 * 9 * 4 * 1024;   // <128,16,1,1>  36864

    // -- encoder (conv bias cancels inside BN, legally dropped) --
    auto run_enc = [&](const float* input, int Cin, const EncP& E,
                       const _Float16* prc1, const _Float16* prc2,
                       const _Float16* pc2, int dstOff) {
        cpinn_conv_direct_kernel<<<8192, 256, 0, stream>>>(input, E.c1w, convout, Cin);
        cpinn_bn_stats_kernel<<<64, 256, 0, stream>>>(convout, 64, meanb, varb);
        cpinn_bn_apply_kernel<<<8192, 256, 0, stream>>>(convout, 64, E.g1, E.b1,
            meanb, varb, actA32, actA16, 64, 0, 0, nullptr);
        cpinn_conv_wmma_t<64, 64, 2, 1><<<512, 256, WB_enc, stream>>>(actA16, prc1, convout);
        cpinn_bn_stats_kernel<<<64, 256, 0, stream>>>(convout, 64, meanb, varb);
        cpinn_bn_apply_kernel<<<8192, 256, 0, stream>>>(convout, 64, E.rg1, E.rb1,
            meanb, varb, nullptr, actB16, 64, 0, 1, nullptr);
        cpinn_conv_wmma_t<64, 64, 2, 1><<<512, 256, WB_enc, stream>>>(actB16, prc2, convout);
        cpinn_bn_stats_kernel<<<64, 256, 0, stream>>>(convout, 64, meanb, varb);
        cpinn_bn_apply_kernel<<<8192, 256, 0, stream>>>(convout, 64, E.rg2, E.rb2,
            meanb, varb, nullptr, actC16, 64, 0, 2, actA32);
        cpinn_conv_wmma_t<64, 64, 2, 1><<<512, 256, WB_enc, stream>>>(actC16, pc2, convout);
        cpinn_bn_stats_kernel<<<64, 256, 0, stream>>>(convout, 64, meanb, varb);
        cpinn_bn_apply_kernel<<<8192, 256, 0, stream>>>(convout, 64, E.g2, E.b2,
            meanb, varb, nullptr, comb_in, 192, dstOff, 0, nullptr);
    };
    run_enc(meteo_in, 2, meteoE, pk_me_rc1, pk_me_rc2, pk_me_c2, 0);   // mf -> ch 0-63
    run_enc(masks_in, 3, maskE,  pk_ma_rc1, pk_ma_rc2, pk_ma_c2, 64);  // kf -> ch 64-127

    // -- out_fn: coord MLP + comb network (mf/kf channels stay fixed) --
    auto run_out = [&](float ex, float ey, float et, float* outbuf) {
        cpinn_mlp_h1_kernel<<<8192, 256, 0, stream>>>(x_in, y_in, t_in, co_w1, co_b1,
                                                      ex, ey, et, h1f16);
        cpinn_conv_wmma_t<64, 64, 2, 0><<<512, 256, WB_mlp, stream>>>(h1f16, pk_w2, convout);
        cpinn_mlp_out_kernel<<<8192, 256, 0, stream>>>(convout, co_b2, comb_in);
        cpinn_conv_wmma_t<192, 128, 1, 1><<<2048, 256, WB_cc1, stream>>>(comb_in, pk_cc1, convout);
        cpinn_bn_stats_kernel<<<128, 256, 0, stream>>>(convout, 128, meanb, varb);
        cpinn_bn_apply_kernel<<<16384, 256, 0, stream>>>(convout, 128, cmb_g1, cmb_b1,
            meanb, varb, actA32, actA16, 128, 0, 0, nullptr);
        cpinn_conv_wmma_t<128, 128, 1, 1><<<2048, 256, WB_crc, stream>>>(actA16, pk_crc1, convout);
        cpinn_bn_stats_kernel<<<128, 256, 0, stream>>>(convout, 128, meanb, varb);
        cpinn_bn_apply_kernel<<<16384, 256, 0, stream>>>(convout, 128, cr_g1, cr_b1,
            meanb, varb, nullptr, actB16, 128, 0, 1, nullptr);
        cpinn_conv_wmma_t<128, 128, 1, 1><<<2048, 256, WB_crc, stream>>>(actB16, pk_crc2, convout);
        cpinn_bn_stats_kernel<<<128, 256, 0, stream>>>(convout, 128, meanb, varb);
        cpinn_bn_apply_kernel<<<16384, 256, 0, stream>>>(convout, 128, cr_g2, cr_b2,
            meanb, varb, nullptr, actC16, 128, 0, 2, actA32);
        cpinn_conv_wmma_t<128, 16, 1, 1><<<256, 256, WB_fin, stream>>>(actC16, pk_cc2, convout);
        cpinn_final_out_kernel<<<384, 256, 0, stream>>>(convout, cmb_c2b, outbuf);
    };

    const float epsS = 0.02f, epsT = 0.02f;
    run_out(0.f, 0.f, 0.f, out_f);          // primal outputs -> d_out[0:98304)
    run_out(+epsS, 0.f, 0.f, oxp);
    run_out(-epsS, 0.f, 0.f, oxm);
    run_out(0.f, +epsS, 0.f, oyp);
    run_out(0.f, -epsS, 0.f, oym);
    run_out(0.f, 0.f, +epsT, otp);
    run_out(0.f, 0.f, -epsT, otm);

    cpinn_zero3_kernel<<<1, 32, 0, stream>>>(accum);
    cpinn_physics_kernel<<<128, 256, 0, stream>>>(out_f, oxp, oxm, oyp, oym, otp, otm,
                                                  log_re, epsS, epsT, accum);
    cpinn_copy3_kernel<<<1, 32, 0, stream>>>(accum, out_f + 98304);
}
